// MetaMixer_6717328851330
// MI455X (gfx1250) — matrix operand order repaired and checked by reading, not compiled, not run on hardware
//
#include <hip/hip_runtime.h>

// ---------------------------------------------------------------------------
// MetaMixer block for MI455X (gfx1250, wave32, WMMA bf16 + async global->LDS)
// ---------------------------------------------------------------------------

typedef __bf16 bf16;
typedef __attribute__((ext_vector_type(16))) __bf16 v16bf;
typedef __attribute__((ext_vector_type(8)))  float  v8f;

#define BDIM 2
#define LDIM 4096
#define HDIM 1024
#define IDIM 2048
#define KTAP 4
#define GRP  8
#define MROWS (BDIM * LDIM)   // 8192

#define BM 128
#define BN 128
#define BK 32
#define LDSS (BK + 8)          // 40 bf16 = 80 B row stride (16B aligned, staggers banks)
#define A_ELEMS (BM * LDSS)    // elements per A stage
#define B_ELEMS (BN * LDSS)
#define LDS_BYTES ((2 * A_ELEMS + 2 * B_ELEMS) * 2)   // 40960 B dynamic LDS

__device__ __forceinline__ float silu_f(float x) {
    return x / (1.0f + __expf(-x));
}

// Per-lane async DMA: 16 B global -> LDS, tracked by ASYNCcnt (gfx1250)
__device__ __forceinline__ void async_b128(unsigned int lds_byte, const bf16* gptr) {
    asm volatile("global_load_async_to_lds_b128 %0, %1, off"
                 :: "v"(lds_byte), "v"((unsigned long long)(uintptr_t)gptr)
                 : "memory");
}
__device__ __forceinline__ void wait_async0() {
    asm volatile("s_wait_asynccnt 0x0" ::: "memory");
}

union Frag { v16bf v; uint4 q[2]; };

// MODE: 0 = split (h_pre / silu(gate)), 1 = grouped-conv im2col GEMM,
//       2 = bias + silu, 3 = bias + residual add, 4 = plain f32 out
template <int MODE>
__global__ __launch_bounds__(256) void gemm_wmma_k(
    const bf16* __restrict__ A, int lda,
    const bf16* __restrict__ Bm, int ldb,
    int Kdim,
    const float* __restrict__ bias,
    const bf16* __restrict__ gate,
    const bf16* __restrict__ resid,
    bf16* __restrict__ out0, bf16* __restrict__ out1,
    float* __restrict__ outf, int ldout)
{
    extern __shared__ bf16 sm[];   // dynamic-only LDS: offset 0 == buffer start
    // layout (elements): [A stage0][A stage1][B stage0][B stage1]

    const int tid     = threadIdx.x;
    const int rowBase = blockIdx.x * BM;
    const int colBase = blockIdx.y * BN;

    // ---- async global -> LDS staging (each thread: one row, two 16B chunks)
    const int lr  = tid >> 1;          // 0..127
    const int lc0 = (tid & 1) * 16;    // 0 or 16

    auto issueStage = [&](int kt, int buf) {
#pragma unroll
        for (int i = 0; i < 2; ++i) {
            const int col  = lc0 + i * 8;
            const unsigned int aIdx = (unsigned)(buf * A_ELEMS + lr * LDSS + col);
            if constexpr (MODE == 1) {
                // im2col gather: packed K index = tap*256 + channel; causal zero-fill
                const int tap    = kt >> 8;                 // 0..3
                const int srcRow = rowBase + lr;
                const int t      = srcRow & (LDIM - 1);
                const int tp     = t + tap - (KTAP - 1);
                if (tp >= 0) {
                    const int gcol = (colBase & ~255) + (kt & 255) + col;
                    async_b128(aIdx * 2u,
                               A + (size_t)(srcRow + tap - (KTAP - 1)) * lda + gcol);
                } else {
                    *(uint4*)&sm[aIdx] = make_uint4(0u, 0u, 0u, 0u);   // ds_store zeros
                }
            } else {
                async_b128(aIdx * 2u, A + (size_t)(rowBase + lr) * lda + kt + col);
            }
            const unsigned int bIdx =
                (unsigned)(2 * A_ELEMS + buf * B_ELEMS + lr * LDSS + col);
            async_b128(bIdx * 2u, Bm + (size_t)(colBase + lr) * ldb + kt + col);
        }
    };

    // ---- wave/lane decomposition: 8 waves, wave tile 64(M) x 32(N)
    const int lane  = tid & 31;
    const int half  = lane >> 4;       // CDNA5 lane-half (K/M interleave)
    const int l15   = lane & 15;
    const int waveM = (tid >> 5) & 1;  // 2 waves along M
    const int waveN = (tid >> 6);      // 4 waves along N

    v8f acc[4][2] = {};

    issueStage(0, 0);

    int cur = 0;
    for (int kt = 0; kt < Kdim; kt += BK) {
        wait_async0();                 // my stage-`cur` DMAs have landed in LDS
        __syncthreads();               // everyone's have (incl. conv zero ds_stores)

        const bool hasNext = (kt + BK) < Kdim;
        if (hasNext) issueStage(kt + BK, cur ^ 1);
        if constexpr (MODE != 1) {
            if (kt + 2 * BK < Kdim) {  // gfx1250 global_prefetch_b8 into L2
                __builtin_prefetch(A + (size_t)(rowBase + lr) * lda + kt + 2 * BK + lc0, 0, 1);
                __builtin_prefetch(Bm + (size_t)(colBase + lr) * ldb + kt + 2 * BK + lc0, 0, 1);
            }
        }

        // ---- LDS -> fragments, per documented CDNA5 WMMA VGPR layouts
        Frag afr[4], bfr[2];
#pragma unroll
        for (int mi = 0; mi < 4; ++mi) {
            const bf16* p = &sm[cur * A_ELEMS + (waveM * 64 + mi * 16 + l15) * LDSS];
            afr[mi].q[0] = *(const uint4*)(p + half * 8);        // K = half*8 .. +7
            afr[mi].q[1] = *(const uint4*)(p + 16 + half * 8);   // K = 16+half*8 .. +7
        }
#pragma unroll
        for (int ni = 0; ni < 2; ++ni) {
            const bf16* p =
                &sm[2 * A_ELEMS + cur * B_ELEMS + (waveN * 32 + ni * 16 + l15) * LDSS];
            bfr[ni].q[0] = *(const uint4*)(p + half * 8);        // K = half*8 .. +7
            bfr[ni].q[1] = *(const uint4*)(p + 16 + half * 8);   // K = 16+half*8 .. +7
        }

#pragma unroll
        for (int mi = 0; mi < 4; ++mi)
#pragma unroll
            for (int ni = 0; ni < 2; ++ni)
                acc[mi][ni] = __builtin_amdgcn_wmma_f32_16x16x32_bf16(
                    false, afr[mi].v, false, bfr[ni].v,
                    (short)0, acc[mi][ni], false, false);

        cur ^= 1;
    }

    // ---- epilogue (C/D layout: VGPR r -> m = r + 8*half, n = lane&15)
#pragma unroll
    for (int mi = 0; mi < 4; ++mi) {
#pragma unroll
        for (int ni = 0; ni < 2; ++ni) {
            const int gcol = colBase + waveN * 32 + ni * 16 + l15;
#pragma unroll
            for (int r = 0; r < 8; ++r) {
                const int grow = rowBase + waveM * 64 + mi * 16 + r + half * 8;
                float x = acc[mi][ni][r];
                if constexpr (MODE == 0) {
                    if (gcol < IDIM)
                        out0[(size_t)grow * ldout + gcol] = (bf16)x;
                    else
                        out1[(size_t)grow * ldout + (gcol - IDIM)] = (bf16)silu_f(x);
                } else if constexpr (MODE == 1) {
                    x += bias[gcol];
                    x *= (float)gate[(size_t)grow * ldout + gcol];
                    out0[(size_t)grow * ldout + gcol] = (bf16)x;
                } else if constexpr (MODE == 2) {
                    out0[(size_t)grow * ldout + gcol] = (bf16)silu_f(x + bias[gcol]);
                } else if constexpr (MODE == 3) {
                    x += bias[gcol] + (float)resid[(size_t)grow * ldout + gcol];
                    out0[(size_t)grow * ldout + gcol] = (bf16)x;
                } else {
                    outf[(size_t)grow * ldout + gcol] = x;
                }
            }
        }
    }
}

// ---------------------------------------------------------------------------
// LayerNorm over IDIM=2048, one block per row, vectorized bf16 I/O
// ---------------------------------------------------------------------------
__global__ __launch_bounds__(256) void layernorm_k(
    const bf16* __restrict__ in, const float* __restrict__ g,
    const float* __restrict__ b, bf16* __restrict__ out)
{
    const int row = blockIdx.x;
    const int tid = threadIdx.x;
    union U { uint4 q; bf16 h[8]; } u, o;
    u.q = *(const uint4*)(in + (size_t)row * IDIM + tid * 8);
    float v[8], s = 0.f, ss = 0.f;
#pragma unroll
    for (int i = 0; i < 8; ++i) { v[i] = (float)u.h[i]; s += v[i]; ss += v[i] * v[i]; }
    __shared__ float sh1[256], sh2[256];
    sh1[tid] = s; sh2[tid] = ss;
    __syncthreads();
    for (int st = 128; st > 0; st >>= 1) {
        if (tid < st) { sh1[tid] += sh1[tid + st]; sh2[tid] += sh2[tid + st]; }
        __syncthreads();
    }
    const float mean = sh1[0] * (1.0f / IDIM);
    const float var  = sh2[0] * (1.0f / IDIM) - mean * mean;
    const float inv  = rsqrtf(var + 1e-5f);
#pragma unroll
    for (int i = 0; i < 8; ++i) {
        const int c = tid * 8 + i;
        o.h[i] = (bf16)((v[i] - mean) * inv * g[c] + b[c]);
    }
    *(uint4*)(out + (size_t)row * IDIM + tid * 8) = o.q;
}

// ---------------------------------------------------------------------------
// f32 -> bf16 conversion + conv-weight repacking ((o,c,k) -> (o, k*256+c))
// ---------------------------------------------------------------------------
__global__ void cvt_bf16_k(const float* __restrict__ in, bf16* __restrict__ out, int n) {
    const int i = blockIdx.x * 256 + threadIdx.x;
    if (i < n) out[i] = (bf16)in[i];
}

__global__ void pack_conv_k(const float* __restrict__ w, bf16* __restrict__ out) {
    const int i = blockIdx.x * 256 + threadIdx.x;   // i = o*1024 + k*256 + c
    const int o   = i >> 10;
    const int rem = i & 1023;
    const int k   = rem >> 8;
    const int c   = rem & 255;
    out[i] = (bf16)w[(size_t)o * 1024 + c * KTAP + k];
}

// ---------------------------------------------------------------------------
extern "C" void kernel_launch(void* const* d_in, const int* in_sizes, int n_in,
                              void* d_out, int out_size, void* d_ws, size_t ws_size,
                              hipStream_t stream)
{
    const float* x        = (const float*)d_in[0];   // (B,L,H)
    const float* in_proj  = (const float*)d_in[1];   // (2I,H)
    const float* conv_w   = (const float*)d_in[2];   // (I,I/G,K)
    const float* conv_b   = (const float*)d_in[3];   // (I)
    const float* ln_g     = (const float*)d_in[4];
    const float* ln_b     = (const float*)d_in[5];
    const float* fc_w     = (const float*)d_in[6];   // (H,I)
    const float* fc_b     = (const float*)d_in[7];   // (H)
    const float* cproj_w  = (const float*)d_in[8];   // (I,H)
    const float* cproj_b  = (const float*)d_in[9];   // (I)
    const float* out_w    = (const float*)d_in[10];  // (H,I)
    float* outp           = (float*)d_out;           // (B,L,H)

    char* ws = (char*)d_ws;
    auto take = [&](size_t elems) {
        char* p = ws;
        ws += (elems * sizeof(bf16) + 255) & ~(size_t)255;
        return (bf16*)p;
    };
    bf16* Xbf    = take((size_t)MROWS * HDIM);
    bf16* Winbf  = take((size_t)2 * IDIM * HDIM);
    bf16* Hpre   = take((size_t)MROWS * IDIM);
    bf16* Gate   = take((size_t)MROWS * IDIM);
    bf16* Wconv  = take((size_t)IDIM * 1024);
    bf16* Resid  = take((size_t)MROWS * IDIM);
    bf16* Hn     = take((size_t)MROWS * IDIM);
    bf16* Wfc    = take((size_t)HDIM * IDIM);
    bf16* T1     = take((size_t)MROWS * HDIM);
    bf16* Wcproj = take((size_t)IDIM * HDIM);
    bf16* H2     = take((size_t)MROWS * IDIM);
    bf16* Wout   = take((size_t)HDIM * IDIM);

    auto cvt = [&](const float* src, bf16* dst, int n) {
        cvt_bf16_k<<<(n + 255) / 256, 256, 0, stream>>>(src, dst, n);
    };
    cvt(x,       Xbf,    MROWS * HDIM);
    cvt(in_proj, Winbf,  2 * IDIM * HDIM);
    cvt(fc_w,    Wfc,    HDIM * IDIM);
    cvt(cproj_w, Wcproj, IDIM * HDIM);
    cvt(out_w,   Wout,   HDIM * IDIM);
    pack_conv_k<<<(IDIM * 1024 + 255) / 256, 256, 0, stream>>>(conv_w, Wconv);

    // GEMM1: p = X @ Win^T ; split -> Hpre, silu -> Gate
    gemm_wmma_k<0><<<dim3(MROWS / BM, (2 * IDIM) / BN), 256, LDS_BYTES, stream>>>(
        Xbf, HDIM, Winbf, HDIM, HDIM,
        nullptr, nullptr, nullptr, Hpre, Gate, nullptr, IDIM);

    // Conv (grouped, causal) as im2col GEMM ; +bias, *gate -> Resid
    gemm_wmma_k<1><<<dim3(MROWS / BM, IDIM / BN), 256, LDS_BYTES, stream>>>(
        Hpre, IDIM, Wconv, 1024, 1024,
        conv_b, Gate, nullptr, Resid, nullptr, nullptr, IDIM);

    // LayerNorm -> Hn
    layernorm_k<<<MROWS, 256, 0, stream>>>(Resid, ln_g, ln_b, Hn);

    // GEMM2: T1 = silu(Hn @ fc_w^T + fc_b)
    gemm_wmma_k<2><<<dim3(MROWS / BM, HDIM / BN), 256, LDS_BYTES, stream>>>(
        Hn, IDIM, Wfc, IDIM, IDIM,
        fc_b, nullptr, nullptr, T1, nullptr, nullptr, HDIM);

    // GEMM3: H2 = T1 @ cproj_w^T + cproj_b + Resid
    gemm_wmma_k<3><<<dim3(MROWS / BM, IDIM / BN), 256, LDS_BYTES, stream>>>(
        T1, HDIM, Wcproj, HDIM, HDIM,
        cproj_b, nullptr, Resid, H2, nullptr, nullptr, IDIM);

    // GEMM4: out = H2 @ out_w^T  (f32)
    gemm_wmma_k<4><<<dim3(MROWS / BM, HDIM / BN), 256, LDS_BYTES, stream>>>(
        H2, IDIM, Wout, IDIM, IDIM,
        nullptr, nullptr, nullptr, nullptr, nullptr, outp, HDIM);

    (void)in_sizes; (void)n_in; (void)out_size; (void)ws_size;
}
